// informer_45964740001913
// MI455X (gfx1250) — compile-verified, matrix operand order checked by reading, not run on hardware
//
#include <hip/hip_runtime.h>
#include <stdint.h>

// ============================================================================
// Informer forward on MI455X (gfx1250, wave32).
// All large matmuls run as bf16 WMMA (v_wmma_f32_16x16x32_bf16, f32 accum).
// LDS tiles are staged by the Tensor Data Mover (tensor_load_to_lds) with
// ping-pong double buffering (TDM fetch of tile k+1 overlaps WMMA on tile k)
// and TDM padding for a conflict-free, 16B-aligned 80-byte LDS row stride.
// Dense attention replaces prob-sparse (gather-hostile, WMMA-friendly).
// ============================================================================

#define B_   8
#define L_   512
#define DM   512
#define NH   8
#define DH_  64
#define DFF  2048
#define CO   7
#define PL   96
#define ME   (B_ * L_)   // 4096 rows
#define BHh  (B_ * NH)   // 64 batched heads

typedef __attribute__((ext_vector_type(16))) __bf16 v16bf;
typedef __attribute__((ext_vector_type(8)))  __bf16 v8bf;
typedef __attribute__((ext_vector_type(8)))  float  v8f;
typedef __attribute__((ext_vector_type(4)))  unsigned int v4u;
typedef __attribute__((ext_vector_type(4)))  int v4i;
typedef __attribute__((ext_vector_type(8)))  int v8i;

#if __has_builtin(__builtin_amdgcn_tensor_load_to_lds) && __has_builtin(__builtin_amdgcn_s_wait_tensorcnt)
#define HAS_TDM 1
#else
#define HAS_TDM 0
#endif

#if HAS_TDM
// Build a D# for a 2D tile: 32 bf16 per row (64B), `rows` rows, global row
// stride `stride_elems`. TDM pads 16B after every 64B stored -> LDS row
// stride 80B (40 bf16): 16B-aligned and LDS-bank conflict free (20-bank row
// stride is 5 four-bank groups, coprime with the 16 groups in 64 banks).
__device__ __forceinline__ void tdm_load_tile_2d(const __bf16* gsrc, unsigned lds_off,
                                                 unsigned rows,
                                                 unsigned long long stride_elems) {
  unsigned long long ga = (unsigned long long)(uintptr_t)gsrc;
  v4u g0;
  g0[0] = 1u;                                             // count=1 (valid)
  g0[1] = lds_off;                                        // LDS byte address
  g0[2] = (unsigned)(ga & 0xFFFFFFFFull);                 // global_addr lo
  g0[3] = (unsigned)((ga >> 32) & 0x1FFFFFFull) | (2u << 30); // addr hi | type=2
  v8i g1;
  g1[0] = (int)((1u << 16) | (1u << 20) | (3u << 22) | (3u << 25));
  //            data_size=2B | pad_en  | pad=64B int. | pad=16B amount
  g1[1] = (int)(32u << 16);                               // tensor_dim0 = 32
  g1[2] = (int)((rows & 0xFFFFu) << 16);                  // tensor_dim1 = rows
  g1[3] = (int)(32u << 16);                               // tile_dim0 = 32
  g1[4] = (int)(rows & 0xFFFFu);                          // tile_dim1 = rows
  g1[5] = (int)(unsigned)(stride_elems & 0xFFFFFFFFull);  // dim0_stride lo32
  g1[6] = (int)(unsigned)((stride_elems >> 32) & 0xFFFFull); // dim0_stride hi16
  g1[7] = 0;
  v4i g2 = {0, 0, 0, 0};
  v4i g3 = {0, 0, 0, 0};
#if defined(__clang_major__) && (__clang_major__ >= 23)
  v8i g4 = {0, 0, 0, 0, 0, 0, 0, 0};
  __builtin_amdgcn_tensor_load_to_lds(g0, g1, g2, g3, g4, 0);
#else
  __builtin_amdgcn_tensor_load_to_lds(g0, g1, g2, g3, 0);
#endif
}
#endif  // HAS_TDM

union Frag16 { v16bf v; v8bf h[2]; };

__device__ __forceinline__ v16bf ld_frag(const __bf16* p) {
  Frag16 f;
  f.h[0] = *(const v8bf*)(p);
  f.h[1] = *(const v8bf*)(p + 8);
  return f.v;
}

// ----------------------------------------------------------------------------
// Big-tile bf16 WMMA GEMM: C[M,N] = act(A[M,K] * BT[N,K]^T + bias + resid)
// Workgroup: 128x128 C tile, 8 waves, each wave 32x64 (2 A-frags x 4 B-frags
// -> 8 WMMAs per K-step, 6 fragment loads). Double-buffered TDM staging:
// waves 0/1 each own one descriptor and wait their own TENSORcnt, so the
// fetch of tile k+1 overlaps the 8 WMMAs on tile k. One barrier per K-step.
// Requires M%128==0, N%128==0, K%32==0.
// ----------------------------------------------------------------------------
__global__ __launch_bounds__(256) void k_gemm_wmma128(
    const __bf16* __restrict__ A, const __bf16* __restrict__ BT,
    const float* __restrict__ bias, const float* __restrict__ resid,
    float* __restrict__ C,
    int M, int N, int K, int lda, int ldb, int ldc,
    long sAb, long sAh, long sBb, long sBh, long sRb, long sRh,
    long sCb, long sCh, int H, int relu) {
  __shared__ __attribute__((aligned(16))) __bf16 sA[2][128 * 40];
  __shared__ __attribute__((aligned(16))) __bf16 sB[2][128 * 40];

  const int z = blockIdx.z, bb = z / H, hh = z % H;
  const __bf16* Ab = A + (size_t)bb * sAb + (size_t)hh * sAh +
                     (size_t)(blockIdx.y * 128) * lda;
  const __bf16* Bb = BT + (size_t)bb * sBb + (size_t)hh * sBh +
                     (size_t)(blockIdx.x * 128) * ldb;

  const int wid = threadIdx.x >> 5, lane = threadIdx.x & 31;
  const int wm = (wid & 3) * 32;        // wave row tile within 128
  const int wn = (wid >> 2) * 64;       // wave col span within 128
  const int lr = lane & 15;
  const int lkh = (lane >> 4) * 16;     // K-half select (A/B frag layout)

  v8f acc[2][4];
#pragma unroll
  for (int r = 0; r < 2; ++r)
#pragma unroll
    for (int c = 0; c < 4; ++c) acc[r][c] = v8f{0.f,0.f,0.f,0.f,0.f,0.f,0.f,0.f};

#if HAS_TDM
  // prologue: fetch tile 0 into buffer 0 (A by wave0, B by wave1)
  if (wid == 0) tdm_load_tile_2d(Ab, (unsigned)(uintptr_t)(const void*)sA[0], 128,
                                 (unsigned long long)lda);
  else if (wid == 1) tdm_load_tile_2d(Bb, (unsigned)(uintptr_t)(const void*)sB[0], 128,
                                      (unsigned long long)ldb);
#endif

  for (int k0 = 0; k0 < K; k0 += 32) {
    const int cur = (k0 >> 5) & 1;
#if HAS_TDM
    if (wid < 2) __builtin_amdgcn_s_wait_tensorcnt(0);
    __syncthreads();  // tile `cur` ready; previous buffer fully consumed
    if (k0 + 32 < K) {
      if (wid == 0)
        tdm_load_tile_2d(Ab + k0 + 32, (unsigned)(uintptr_t)(const void*)sA[cur ^ 1],
                         128, (unsigned long long)lda);
      else if (wid == 1)
        tdm_load_tile_2d(Bb + k0 + 32, (unsigned)(uintptr_t)(const void*)sB[cur ^ 1],
                         128, (unsigned long long)ldb);
    }
#else
    __syncthreads();
    for (int i = threadIdx.x; i < 128 * 32; i += 256) {
      int r = i >> 5, c = i & 31;
      sA[cur][r * 40 + c] = Ab[(size_t)r * lda + k0 + c];
      sB[cur][r * 40 + c] = Bb[(size_t)r * ldb + k0 + c];
    }
    __syncthreads();
#endif
    if (k0 + 64 < K) {  // L2 hint for the tile after the in-flight one
      __builtin_prefetch(Ab + (size_t)(wm + lr) * lda + k0 + 64, 0, 1);
      __builtin_prefetch(Bb + (size_t)(wn + lr) * ldb + k0 + 64, 0, 1);
    }
    v16bf a0 = ld_frag(&sA[cur][(wm + lr) * 40 + lkh]);
    v16bf a1 = ld_frag(&sA[cur][(wm + 16 + lr) * 40 + lkh]);
#pragma unroll
    for (int c = 0; c < 4; ++c) {
      v16bf bfc = ld_frag(&sB[cur][(wn + c * 16 + lr) * 40 + lkh]);
      acc[0][c] = __builtin_amdgcn_wmma_f32_16x16x32_bf16(false, a0, false, bfc,
                                                          (short)0, acc[0][c],
                                                          false, false);
      acc[1][c] = __builtin_amdgcn_wmma_f32_16x16x32_bf16(false, a1, false, bfc,
                                                          (short)0, acc[1][c],
                                                          false, false);
    }
  }

  // Epilogue per C layout: VGPR i <-> row i + 8*(lane>=16); col = lane&15.
  const int gr_base = blockIdx.y * 128 + wm + (lane >> 4) * 8;
  const int gc_base = blockIdx.x * 128 + wn + lr;
  float* Cb = C + (size_t)bb * sCb + (size_t)hh * sCh;
  const float* Rb = resid ? resid + (size_t)bb * sRb + (size_t)hh * sRh : nullptr;
#pragma unroll
  for (int rt = 0; rt < 2; ++rt) {
#pragma unroll
    for (int ct = 0; ct < 4; ++ct) {
      const int gc = gc_base + ct * 16;
      const float bi = bias ? bias[gc] : 0.f;
#pragma unroll
      for (int i = 0; i < 8; ++i) {
        int gr = gr_base + rt * 16 + i;
        float x = acc[rt][ct][i] + bi;
        size_t o = (size_t)gr * ldc + gc;
        if (Rb) x += Rb[o];
        if (relu) x = fmaxf(x, 0.f);
        Cb[o] = x;
      }
    }
  }
}

// ----------------------------------------------------------------------------
// 64x64 bf16 WMMA GEMM (for N=64 context GEMM); same design, wave tile 16x32,
// double-buffered TDM staging.
// ----------------------------------------------------------------------------
__global__ __launch_bounds__(256) void k_gemm_wmma64(
    const __bf16* __restrict__ A, const __bf16* __restrict__ BT,
    const float* __restrict__ bias, const float* __restrict__ resid,
    float* __restrict__ C,
    int M, int N, int K, int lda, int ldb, int ldc,
    long sAb, long sAh, long sBb, long sBh, long sRb, long sRh,
    long sCb, long sCh, int H, int relu) {
  __shared__ __attribute__((aligned(16))) __bf16 sA[2][64 * 40];
  __shared__ __attribute__((aligned(16))) __bf16 sB[2][64 * 40];

  const int z = blockIdx.z, bb = z / H, hh = z % H;
  const __bf16* Ab = A + (size_t)bb * sAb + (size_t)hh * sAh +
                     (size_t)(blockIdx.y * 64) * lda;
  const __bf16* Bb = BT + (size_t)bb * sBb + (size_t)hh * sBh +
                     (size_t)(blockIdx.x * 64) * ldb;

  const int wid = threadIdx.x >> 5, lane = threadIdx.x & 31;
  const int wm = (wid & 3) * 16;
  const int wn = (wid >> 2) * 32;
  const int lr = lane & 15;
  const int lkh = (lane >> 4) * 16;

  v8f acc0 = {0.f, 0.f, 0.f, 0.f, 0.f, 0.f, 0.f, 0.f};
  v8f acc1 = {0.f, 0.f, 0.f, 0.f, 0.f, 0.f, 0.f, 0.f};

#if HAS_TDM
  if (wid == 0) tdm_load_tile_2d(Ab, (unsigned)(uintptr_t)(const void*)sA[0], 64,
                                 (unsigned long long)lda);
  else if (wid == 1) tdm_load_tile_2d(Bb, (unsigned)(uintptr_t)(const void*)sB[0], 64,
                                      (unsigned long long)ldb);
#endif

  for (int k0 = 0; k0 < K; k0 += 32) {
    const int cur = (k0 >> 5) & 1;
#if HAS_TDM
    if (wid < 2) __builtin_amdgcn_s_wait_tensorcnt(0);
    __syncthreads();
    if (k0 + 32 < K) {
      if (wid == 0)
        tdm_load_tile_2d(Ab + k0 + 32, (unsigned)(uintptr_t)(const void*)sA[cur ^ 1],
                         64, (unsigned long long)lda);
      else if (wid == 1)
        tdm_load_tile_2d(Bb + k0 + 32, (unsigned)(uintptr_t)(const void*)sB[cur ^ 1],
                         64, (unsigned long long)ldb);
    }
#else
    __syncthreads();
    for (int i = threadIdx.x; i < 64 * 32; i += 256) {
      int r = i >> 5, c = i & 31;
      sA[cur][r * 40 + c] = Ab[(size_t)r * lda + k0 + c];
      sB[cur][r * 40 + c] = Bb[(size_t)r * ldb + k0 + c];
    }
    __syncthreads();
#endif
    if (k0 + 64 < K) {
      __builtin_prefetch(Ab + (size_t)(wm + lr) * lda + k0 + 64, 0, 1);
      __builtin_prefetch(Bb + (size_t)(wn + lr) * ldb + k0 + 64, 0, 1);
    }
    v16bf af  = ld_frag(&sA[cur][(wm + lr) * 40 + lkh]);
    v16bf bf0 = ld_frag(&sB[cur][(wn + lr) * 40 + lkh]);
    v16bf bf1 = ld_frag(&sB[cur][(wn + 16 + lr) * 40 + lkh]);
    acc0 = __builtin_amdgcn_wmma_f32_16x16x32_bf16(false, af, false, bf0,
                                                   (short)0, acc0, false, false);
    acc1 = __builtin_amdgcn_wmma_f32_16x16x32_bf16(false, af, false, bf1,
                                                   (short)0, acc1, false, false);
  }

  const int gr0 = blockIdx.y * 64 + wm + (lane >> 4) * 8;
  const int gc0 = blockIdx.x * 64 + wn + lr;
  float* Cb = C + (size_t)bb * sCb + (size_t)hh * sCh;
  const float* Rb = resid ? resid + (size_t)bb * sRb + (size_t)hh * sRh : nullptr;
#pragma unroll
  for (int i = 0; i < 8; ++i) {
    int gr = gr0 + i;
    float x0 = acc0[i], x1 = acc1[i];
    if (bias) { x0 += bias[gc0]; x1 += bias[gc0 + 16]; }
    size_t o = (size_t)gr * ldc + gc0;
    if (Rb) { x0 += Rb[o]; x1 += Rb[o + 16]; }
    if (relu) { x0 = fmaxf(x0, 0.f); x1 = fmaxf(x1, 0.f); }
    Cb[o] = x0;
    Cb[o + 16] = x1;
  }
}

// ---------------------------------------------------------------------------
// Elementwise / reduction kernels
// ---------------------------------------------------------------------------
__device__ __forceinline__ float block_sum(float v, float* red) {
  int t = threadIdx.x;
  red[t] = v; __syncthreads();
  for (int o = 128; o > 0; o >>= 1) {
    if (t < o) red[t] += red[t + o];
    __syncthreads();
  }
  float r = red[0]; __syncthreads();
  return r;
}
__device__ __forceinline__ float block_max(float v, float* red) {
  int t = threadIdx.x;
  red[t] = v; __syncthreads();
  for (int o = 128; o > 0; o >>= 1) {
    if (t < o) red[t] = fmaxf(red[t], red[t + o]);
    __syncthreads();
  }
  float r = red[0]; __syncthreads();
  return r;
}

// token conv (k=3, circular pad) + sinusoidal position embedding
__global__ __launch_bounds__(256) void k_embed(const float* __restrict__ x,
                                               const float* __restrict__ w,
                                               float* __restrict__ h) {
  int idx = blockIdx.x * 256 + threadIdx.x;
  if (idx >= ME * DM) return;
  int d = idx & (DM - 1);
  int l = (idx >> 9) & (L_ - 1);
  int b = idx >> 18;
  int r0 = (l == 0) ? (L_ - 1) : (l - 1);
  int r2 = (l == L_ - 1) ? 0 : (l + 1);
  int rr[3] = {r0, l, r2};
  float acc = 0.f;
#pragma unroll
  for (int t = 0; t < 3; ++t) {
    const float* xr = x + ((size_t)b * L_ + rr[t]) * CO;
    const float* wr = w + (size_t)t * CO * DM + d;
#pragma unroll
    for (int c = 0; c < CO; ++c) acc += xr[c] * wr[(size_t)c * DM];
  }
  int i2 = d >> 1;
  float ang = (float)l * __expf(-(float)(2 * i2) * (9.2103403719761836f / (float)DM));
  acc += (d & 1) ? __cosf(ang) : __sinf(ang);
  h[idx] = acc;
}

__global__ __launch_bounds__(256) void k_layernorm(const float* __restrict__ x,
                                                   const float* __restrict__ g,
                                                   const float* __restrict__ b,
                                                   float* __restrict__ y, int D) {
  __shared__ float red[256];
  const float* xr = x + (size_t)blockIdx.x * D;
  float* yr = y + (size_t)blockIdx.x * D;
  float s = 0.f;
  for (int c = threadIdx.x; c < D; c += 256) s += xr[c];
  float mu = block_sum(s, red) / (float)D;
  float v = 0.f;
  for (int c = threadIdx.x; c < D; c += 256) { float d = xr[c] - mu; v += d * d; }
  float var = block_sum(v, red) / (float)D;
  float inv = rsqrtf(var + 1e-5f);
  for (int c = threadIdx.x; c < D; c += 256)
    yr[c] = (xr[c] - mu) * inv * g[c] + b[c];
}

__global__ __launch_bounds__(256) void k_softmax(float* __restrict__ s, int Lk,
                                                 int Lq, float scale, int causal) {
  __shared__ float red[256];
  int q = (int)(blockIdx.x % (unsigned)Lq);
  float* r = s + (size_t)blockIdx.x * Lk;
  float mx = -3.0e38f;
  for (int c = threadIdx.x; c < Lk; c += 256) {
    if (causal && c > q) continue;
    mx = fmaxf(mx, r[c] * scale);
  }
  mx = block_max(mx, red);
  float sum = 0.f;
  for (int c = threadIdx.x; c < Lk; c += 256) {
    float e = (causal && c > q) ? 0.f : __expf(r[c] * scale - mx);
    r[c] = e; sum += e;
  }
  sum = block_sum(sum, red);
  float inv = 1.f / sum;
  for (int c = threadIdx.x; c < Lk; c += 256) r[c] *= inv;
}

__global__ __launch_bounds__(256) void k_cvt(const float* __restrict__ s,
                                             __bf16* __restrict__ d, long n) {
  long i = (long)blockIdx.x * 256 + threadIdx.x;
  if (i < n) d[i] = (__bf16)s[i];
}

// W[R,C] f32 -> W^T[C,R] bf16 (weights stored N-major for the GEMM B operand)
__global__ __launch_bounds__(256) void k_cvt_tr(const float* __restrict__ s,
                                                __bf16* __restrict__ d, int R, int C) {
  long i = (long)blockIdx.x * 256 + threadIdx.x;
  if (i >= (long)R * C) return;
  int c = (int)(i % C);
  int r = (int)(i / C);
  d[(size_t)c * R + r] = (__bf16)s[i];
}

// V[B*L, H*Dh] f32 -> Vt[B,H,Dh,L] bf16 (per-head V^T for ctx GEMM)
__global__ __launch_bounds__(256) void k_cvt_tr_heads(const float* __restrict__ s,
                                                      __bf16* __restrict__ d) {
  int i = blockIdx.x * 256 + threadIdx.x;
  if (i >= ME * DM) return;
  int col = i & (DM - 1);
  int row = i >> 9;
  int b = row >> 9, l = row & (L_ - 1);
  int h = col >> 6, dh = col & (DH_ - 1);
  d[(((size_t)(b * NH + h) * DH_ + dh) * L_) + l] = (__bf16)s[i];
}

__global__ __launch_bounds__(256) void k_dec_proj(const float* __restrict__ x,
                                                  const float* __restrict__ w,
                                                  const float* __restrict__ bias,
                                                  float* __restrict__ y) {
  int i = blockIdx.x * 256 + threadIdx.x;
  if (i >= ME * CO) return;
  int n = i % CO, m = i / CO;
  float a = bias[n];
  for (int k = 0; k < DM; ++k) a += x[(size_t)m * DM + k] * w[(size_t)k * CO + n];
  y[i] = a;
}

__global__ __launch_bounds__(256) void k_time_proj(const float* __restrict__ dp,
                                                   const float* __restrict__ tw,
                                                   const float* __restrict__ tb,
                                                   float* __restrict__ out) {
  int i = blockIdx.x * 256 + threadIdx.x;
  if (i >= B_ * PL * CO) return;
  int c = i % CO, p = (i / CO) % PL, b = i / (CO * PL);
  float a = tb[p];
  for (int l = 0; l < L_; ++l)
    a += dp[((size_t)b * L_ + l) * CO + c] * tw[(size_t)l * PL + p];
  out[i] = a;
}

// ---------------------------------------------------------------------------
// Host-side orchestration
// ---------------------------------------------------------------------------
struct AttnP { const float *qw, *qb, *kw, *kb, *vw, *vb, *ow, *ob; };

struct WS {
  float *h, *x2, *q, *k, *v, *cat, *enc_out, *ffy, *scores, *dproj;
  __bf16 *abf, *ybf, *qbf, *kbf, *vtbf, *sbf, *wbf;
};

static inline void gemmL(hipStream_t st, const __bf16* A, const __bf16* BT,
                         const float* bias, const float* resid, float* C,
                         int M, int N, int K, int lda, int ldb, int ldc,
                         long sAb, long sAh, long sBb, long sBh, long sRb, long sRh,
                         long sCb, long sCh, int batches, int H, int relu) {
  if ((M % 128) == 0 && (N % 128) == 0) {
    dim3 g((unsigned)(N / 128), (unsigned)(M / 128), (unsigned)batches);
    k_gemm_wmma128<<<g, 256, 0, st>>>(A, BT, bias, resid, C, M, N, K, lda, ldb,
                                      ldc, sAb, sAh, sBb, sBh, sRb, sRh, sCb,
                                      sCh, H, relu);
  } else {
    dim3 g((unsigned)(N / 64), (unsigned)(M / 64), (unsigned)batches);
    k_gemm_wmma64<<<g, 256, 0, st>>>(A, BT, bias, resid, C, M, N, K, lda, ldb,
                                     ldc, sAb, sAh, sBb, sBh, sRb, sRh, sCb,
                                     sCh, H, relu);
  }
}
static inline void gemm_lin(hipStream_t st, const __bf16* A, const __bf16* WT,
                            const float* bias, const float* resid, float* C,
                            int M, int N, int K, int relu) {
  gemmL(st, A, WT, bias, resid, C, M, N, K, K, K, N,
        0, 0, 0, 0, 0, 0, 0, 0, 1, 1, relu);
}
static inline void cvt(hipStream_t st, const float* s, __bf16* d, long n) {
  k_cvt<<<(unsigned)((n + 255) / 256), 256, 0, st>>>(s, d, n);
}
static inline void cvt_tr(hipStream_t st, const float* s, __bf16* d, int R, int C) {
  long n = (long)R * C;
  k_cvt_tr<<<(unsigned)((n + 255) / 256), 256, 0, st>>>(s, d, R, C);
}
static inline void ln(hipStream_t st, const float* x, const float* g,
                      const float* b, float* y) {
  k_layernorm<<<ME, 256, 0, st>>>(x, g, b, y, DM);
}

// Dense multi-head attention, all GEMMs on WMMA; O-proj fuses residual(xq).
static void run_attn(hipStream_t st, WS& S, const float* xq, const float* xkv,
                     const AttnP& P, float* out, int causal) {
  cvt(st, xq, S.abf, (long)ME * DM);
  cvt_tr(st, P.qw, S.wbf, DM, DM);
  gemm_lin(st, S.abf, S.wbf, P.qb, nullptr, S.q, ME, DM, DM, 0);
  if (xkv != xq) cvt(st, xkv, S.abf, (long)ME * DM);
  cvt_tr(st, P.kw, S.wbf, DM, DM);
  gemm_lin(st, S.abf, S.wbf, P.kb, nullptr, S.k, ME, DM, DM, 0);
  cvt_tr(st, P.vw, S.wbf, DM, DM);
  gemm_lin(st, S.abf, S.wbf, P.vb, nullptr, S.v, ME, DM, DM, 0);

  cvt(st, S.q, S.qbf, (long)ME * DM);
  cvt(st, S.k, S.kbf, (long)ME * DM);
  k_cvt_tr_heads<<<(ME * DM + 255) / 256, 256, 0, st>>>(S.v, S.vtbf);

  // scores[b,h] = Q_bh @ K_bh^T   (M=L, N=L, K=Dh; strided head views)
  gemmL(st, S.qbf, S.kbf, nullptr, nullptr, S.scores,
        L_, L_, DH_, DM, DM, L_,
        (long)L_ * DM, DH_, (long)L_ * DM, DH_, 0, 0,
        (long)NH * L_ * L_, (long)L_ * L_, BHh, NH, 0);
  k_softmax<<<BHh * L_, 256, 0, st>>>(S.scores, L_, L_, 0.125f, causal);
  cvt(st, S.scores, S.sbf, (long)BHh * L_ * L_);

  // ctx[b,h] = P @ V_bh, written interleaved back to [B*L, H*Dh]  (N=64 path)
  gemmL(st, S.sbf, S.vtbf, nullptr, nullptr, S.cat,
        L_, DH_, L_, L_, L_, DM,
        (long)NH * L_ * L_, (long)L_ * L_, (long)NH * DH_ * L_, (long)DH_ * L_, 0, 0,
        (long)L_ * DM, DH_, BHh, NH, 0);

  cvt(st, S.cat, S.abf, (long)ME * DM);
  cvt_tr(st, P.ow, S.wbf, DM, DM);
  gemm_lin(st, S.abf, S.wbf, P.ob, xq, out, ME, DM, DM, 0);
}

static void run_ff(hipStream_t st, WS& S, const float* x, const float* w1,
                   const float* b1, const float* w2, const float* b2, float* out) {
  cvt(st, x, S.abf, (long)ME * DM);
  cvt_tr(st, w1, S.wbf, DM, DFF);
  gemm_lin(st, S.abf, S.wbf, b1, nullptr, S.ffy, ME, DFF, DM, 1);   // fused ReLU
  cvt(st, S.ffy, S.ybf, (long)ME * DFF);
  cvt_tr(st, w2, S.wbf, DFF, DM);
  gemm_lin(st, S.ybf, S.wbf, b2, x, out, ME, DM, DFF, 0);           // fused resid
}

// Input leaf order assumes jax pytree (sorted dict keys) flattening.
enum {
  D_CROSS_KB = 0, D_CROSS_KW, D_CROSS_OB, D_CROSS_OW,
  D_CROSS_QB, D_CROSS_QW, D_CROSS_VB, D_CROSS_VW,
  D_FF1B, D_FF1W, D_FF2B, D_FF2W,
  D_LN1B, D_LN1G, D_LN2B, D_LN2G, D_LN3B, D_LN3G,
  D_SELF_KB, D_SELF_KW, D_SELF_OB, D_SELF_OW,
  D_SELF_QB, D_SELF_QW, D_SELF_VB, D_SELF_VW,
  I_DNORM_B = 26, I_DNORM_G, I_DPROJ_B, I_DPROJ_W,
  I_ENC0 = 30, I_ENC1 = 46,
  I_ENORM_B = 62, I_ENORM_G,
  I_TIME_B = 64, I_TIME_W,
  I_TOKCONV = 66, I_XENC = 67, I_NLEAVES = 68
};
// within an encoder layer (16 leaves):
// 0 k.b 1 k.w 2 o.b 3 o.w 4 q.b 5 q.w 6 v.b 7 v.w
// 8 ff1.b 9 ff1.w 10 ff2.b 11 ff2.w 12 ln1.b 13 ln1.g 14 ln2.b 15 ln2.g

extern "C" void kernel_launch(void* const* d_in, const int* in_sizes, int n_in,
                              void* d_out, int out_size, void* d_ws, size_t ws_size,
                              hipStream_t stream) {
  if (n_in < I_NLEAVES) return;
  const float* in[I_NLEAVES];
  for (int i = 0; i < I_NLEAVES; ++i) in[i] = (const float*)d_in[i];

  char* base = (char*)d_ws;
  size_t off = 0;
  auto take = [&](size_t bytes) {
    char* p = base + off;
    off = (off + bytes + 255) & ~(size_t)255;
    return (void*)p;
  };
  WS S;
  S.h       = (float*)take((size_t)ME * DM * 4);
  S.x2      = (float*)take((size_t)ME * DM * 4);
  S.q       = (float*)take((size_t)ME * DM * 4);
  S.k       = (float*)take((size_t)ME * DM * 4);
  S.v       = (float*)take((size_t)ME * DM * 4);
  S.cat     = (float*)take((size_t)ME * DM * 4);
  S.enc_out = (float*)take((size_t)ME * DM * 4);
  S.ffy     = (float*)take((size_t)ME * DFF * 4);
  S.scores  = (float*)take((size_t)BHh * L_ * L_ * 4);
  S.dproj   = (float*)take((size_t)ME * CO * 4);
  S.abf     = (__bf16*)take((size_t)ME * DM * 2);
  S.ybf     = (__bf16*)take((size_t)ME * DFF * 2);
  S.qbf     = (__bf16*)take((size_t)ME * DM * 2);
  S.kbf     = (__bf16*)take((size_t)ME * DM * 2);
  S.vtbf    = (__bf16*)take((size_t)ME * DM * 2);
  S.sbf     = (__bf16*)take((size_t)BHh * L_ * L_ * 2);
  S.wbf     = (__bf16*)take((size_t)DM * DFF * 2);
  (void)ws_size; (void)in_sizes;

  // --- embedding: conv3 + positional encoding ---
  k_embed<<<(ME * DM + 255) / 256, 256, 0, stream>>>(in[I_XENC], in[I_TOKCONV], S.h);

  // --- encoder layers ---
  for (int li = 0; li < 2; ++li) {
    int e = (li == 0) ? I_ENC0 : I_ENC1;
    AttnP ap = {in[e + 5], in[e + 4], in[e + 1], in[e + 0],
                in[e + 7], in[e + 6], in[e + 3], in[e + 2]};
    run_attn(stream, S, S.h, S.h, ap, S.x2, 1);
    ln(stream, S.x2, in[e + 13], in[e + 12], S.h);
    run_ff(stream, S, S.h, in[e + 9], in[e + 8], in[e + 11], in[e + 10], S.x2);
    ln(stream, S.x2, in[e + 15], in[e + 14], S.h);
  }
  ln(stream, S.h, in[I_ENORM_G], in[I_ENORM_B], S.enc_out);

  // --- decoder layer ---
  hipMemcpyAsync(S.h, S.enc_out, (size_t)ME * DM * 4,
                 hipMemcpyDeviceToDevice, stream);
  {
    AttnP self = {in[D_SELF_QW], in[D_SELF_QB], in[D_SELF_KW], in[D_SELF_KB],
                  in[D_SELF_VW], in[D_SELF_VB], in[D_SELF_OW], in[D_SELF_OB]};
    AttnP cross = {in[D_CROSS_QW], in[D_CROSS_QB], in[D_CROSS_KW], in[D_CROSS_KB],
                   in[D_CROSS_VW], in[D_CROSS_VB], in[D_CROSS_OW], in[D_CROSS_OB]};
    run_attn(stream, S, S.h, S.h, self, S.x2, 1);
    ln(stream, S.x2, in[D_LN1G], in[D_LN1B], S.h);
    run_attn(stream, S, S.h, S.enc_out, cross, S.x2, 1);
    ln(stream, S.x2, in[D_LN2G], in[D_LN2B], S.h);
    run_ff(stream, S, S.h, in[D_FF1W], in[D_FF1B], in[D_FF2W], in[D_FF2B], S.x2);
    ln(stream, S.x2, in[D_LN3G], in[D_LN3B], S.h);
  }
  ln(stream, S.h, in[I_DNORM_G], in[I_DNORM_B], S.x2);

  // --- output projections ---
  k_dec_proj<<<(ME * CO + 255) / 256, 256, 0, stream>>>(S.x2, in[I_DPROJ_W],
                                                        in[I_DPROJ_B], S.dproj);
  k_time_proj<<<(B_ * PL * CO + 255) / 256, 256, 0, stream>>>(
      S.dproj, in[I_TIME_W], in[I_TIME_B], (float*)d_out);
  (void)out_size;
}
